// RNNPointerGenerator_32959579030358
// MI455X (gfx1250) — matrix operand order repaired and checked by reading, not compiled, not run on hardware
//
#include <hip/hip_runtime.h>
#include <hip/hip_bf16.h>
#include <cstdint>

#define B_   256
#define S_   512
#define H_   512
#define E_   512
#define V_   50257
#define ENC_ 1024
#define FEAT_ 2048   // 3H + E
#define EPS_ 1e-40f

typedef __attribute__((ext_vector_type(16))) _Float16 v16h;
typedef __attribute__((ext_vector_type(8)))  float    v8f;

// ---------------------------------------------------------------------------
// k1: context_state (LDS) + fused p_gen.  One block per batch row b.
// Streams 512 MB of encoder_output -> bandwidth bound part of the problem.
// ---------------------------------------------------------------------------
__global__ __launch_bounds__(256) void k1_context_pgen(
    const float* __restrict__ enc,        // [S,B,ENC]
    const float* __restrict__ attn,       // [B,S]
    const float* __restrict__ hidden,     // [B,H]
    const float* __restrict__ embed,      // [B,E]
    const float* __restrict__ pgen_w,     // [1, FEAT]
    const float* __restrict__ pgen_b,     // [1]
    float* __restrict__ pgen_out)         // [B]
{
    __shared__ float ctx[ENC_];
    __shared__ float red[256];
    const int b = blockIdx.x;
    const int t = threadIdx.x;

    float a0 = 0.f, a1 = 0.f, a2 = 0.f, a3 = 0.f;
    const float* attn_b = attn + (size_t)b * S_;
    for (int s = 0; s < S_; ++s) {
        const float a = attn_b[s];
        const float* p = enc + (size_t)s * (B_ * ENC_) + (size_t)b * ENC_;
        a0 += a * p[t];
        a1 += a * p[t + 256];
        a2 += a * p[t + 512];
        a3 += a * p[t + 768];
    }
    ctx[t]       = a0;
    ctx[t + 256] = a1;
    ctx[t + 512] = a2;
    ctx[t + 768] = a3;
    __syncthreads();

    // feat = [hidden(512) | embed(512) | ctx(1024)] dot pgen_w
    float part = 0.f;
    for (int j = t; j < FEAT_; j += 256) {
        float f;
        if (j < H_)            f = hidden[(size_t)b * H_ + j];
        else if (j < H_ + E_)  f = embed[(size_t)b * E_ + (j - H_)];
        else                   f = ctx[j - (H_ + E_)];
        part += f * pgen_w[j];
    }
    red[t] = part;
    __syncthreads();
    for (int off = 128; off > 0; off >>= 1) {
        if (t < off) red[t] += red[t + off];
        __syncthreads();
    }
    if (t == 0) {
        const float z = red[0] + pgen_b[0];
        pgen_out[b] = 1.0f / (1.0f + __expf(-z));
    }
}

// ---------------------------------------------------------------------------
// k2: logits = x @ proj_w.T + proj_b via v_wmma_f32_16x16x32_f16.
// Block = 8 waves (wave32). Each wave owns one 16(N)-col tile; block tile is
// 16 rows x 128 cols. x tile staged in LDS as f16 (padded stride -> conflict
// free ds_load_b128). proj_w rows converted f32->f16 per lane (hits L2 after
// first touch; 103 MB < 192 MB L2).
// ---------------------------------------------------------------------------
#define AS_STRIDE 520   // 512 + 8 halves pad: 16B-aligned rows, bank-conflict free

__global__ __launch_bounds__(256) void k2_gemm_wmma(
    const float* __restrict__ x,        // [B,H]
    const float* __restrict__ proj_w,   // [V,H]
    const float* __restrict__ proj_b,   // [V]
    float* __restrict__ logits)         // [B,V]
{
    __shared__ __align__(16) _Float16 As[16 * AS_STRIDE];

    const int mbase = blockIdx.y * 16;     // batch-row tile
    const int vbase = blockIdx.x * 128;    // vocab-col tile
    const int wave  = threadIdx.x >> 5;
    const int lane  = threadIdx.x & 31;

    // Stage A tile (16 x 512) as f16 into LDS
    for (int idx = threadIdx.x; idx < 16 * H_; idx += 256) {
        const int m = idx >> 9;          // /512
        const int k = idx & 511;
        As[m * AS_STRIDE + k] = (_Float16)x[(size_t)(mbase + m) * H_ + k];
    }
    __syncthreads();

    const int  vcol = vbase + wave * 16 + (lane & 15);
    const bool vok  = vcol < V_;
    const float* wrow = proj_w + (size_t)(vok ? vcol : 0) * H_;

    v8f c = {};
    const int m    = lane & 15;
    const int koff = (lane < 16) ? 0 : 16;  // A-matrix K sub-block per half-wave
    const int kbof = (lane < 16) ? 0 : 16;  // B-matrix K sub-block per half-wave

#pragma unroll 4
    for (int kk = 0; kk < H_; kk += 32) {
        // ----- A fragment: row m, K = kk + koff + 0..15
        union { v16h v; float4 f4[2]; } afrag;
        afrag.f4[0] = *(const float4*)&As[m * AS_STRIDE + kk + koff];
        afrag.f4[1] = *(const float4*)&As[m * AS_STRIDE + kk + koff + 8];

        // ----- B fragment: col vcol, K = kk + kbof + 0..15 (f32 -> f16)
        const float4 w0 = *(const float4*)&wrow[kk + kbof];
        const float4 w1 = *(const float4*)&wrow[kk + kbof + 4];
        const float4 w2 = *(const float4*)&wrow[kk + kbof + 8];
        const float4 w3 = *(const float4*)&wrow[kk + kbof + 12];
        union { v16h v; _Float16 h[16]; } bfrag;
        bfrag.h[0]  = (_Float16)w0.x; bfrag.h[1]  = (_Float16)w0.y;
        bfrag.h[2]  = (_Float16)w0.z; bfrag.h[3]  = (_Float16)w0.w;
        bfrag.h[4]  = (_Float16)w1.x; bfrag.h[5]  = (_Float16)w1.y;
        bfrag.h[6]  = (_Float16)w1.z; bfrag.h[7]  = (_Float16)w1.w;
        bfrag.h[8]  = (_Float16)w2.x; bfrag.h[9]  = (_Float16)w2.y;
        bfrag.h[10] = (_Float16)w2.z; bfrag.h[11] = (_Float16)w2.w;
        bfrag.h[12] = (_Float16)w3.x; bfrag.h[13] = (_Float16)w3.y;
        bfrag.h[14] = (_Float16)w3.z; bfrag.h[15] = (_Float16)w3.w;

        c = __builtin_amdgcn_wmma_f32_16x16x32_f16(
                /*neg_a=*/false, afrag.v, /*neg_b=*/false, bfrag.v,
                /*c_mod=*/(short)0, c, /*reuse_a=*/false, /*reuse_b=*/false);
    }

    // Epilogue: D layout — VGPR j: M = j (+8 for lanes 16..31), N = lane&15
    if (vok) {
        const float bias = proj_b[vcol];
        const int mrow = (lane < 16) ? 0 : 8;
#pragma unroll
        for (int j = 0; j < 8; ++j) {
            logits[(size_t)(mbase + mrow + j) * V_ + vcol] = c[j] + bias;
        }
    }
}

// ---------------------------------------------------------------------------
// k3: per-row online softmax stats (max, sum of exp). One block per row.
// ---------------------------------------------------------------------------
__global__ __launch_bounds__(256) void k3_rowstats(
    const float* __restrict__ logits,   // [B,V]
    float* __restrict__ row_m,          // [B]
    float* __restrict__ row_s)          // [B]
{
    __shared__ float sm[256];
    __shared__ float ss[256];
    const int b = blockIdx.x;
    const int t = threadIdx.x;
    const float* row = logits + (size_t)b * V_;

    float m = -3.402823e38f, s = 0.f;
    for (int v = t; v < V_; v += 256) {
        const float xv = row[v];
        const float nm = fmaxf(m, xv);
        s = s * __expf(m - nm) + __expf(xv - nm);
        m = nm;
    }
    sm[t] = m; ss[t] = s;
    __syncthreads();
    for (int off = 128; off > 0; off >>= 1) {
        if (t < off) {
            const float m2 = sm[t + off], s2 = ss[t + off];
            const float nm = fmaxf(sm[t], m2);
            ss[t] = ss[t] * __expf(sm[t] - nm) + s2 * __expf(m2 - nm);
            sm[t] = nm;
        }
        __syncthreads();
    }
    if (t == 0) { row_m[b] = sm[0]; row_s[b] = ss[0]; }
}

// ---------------------------------------------------------------------------
// k4: in-place  logits -> p_gen * softmax(logits)
// ---------------------------------------------------------------------------
__global__ __launch_bounds__(256) void k4_mix(
    float* __restrict__ buf,            // [B,V] in-place
    const float* __restrict__ pgen,
    const float* __restrict__ row_m,
    const float* __restrict__ row_s)
{
    const size_t total = (size_t)B_ * V_;
    for (size_t i = (size_t)blockIdx.x * blockDim.x + threadIdx.x;
         i < total; i += (size_t)gridDim.x * blockDim.x) {
        const int b = (int)(i / V_);
        const float scale = pgen[b] / row_s[b];
        buf[i] = scale * __expf(buf[i] - row_m[b]);
    }
}

// ---------------------------------------------------------------------------
// k5: pointer scatter: buf[b, ids[b,s]] += (1-p_gen[b]) * attn[b,s]
// ---------------------------------------------------------------------------
__global__ __launch_bounds__(256) void k5_scatter(
    float* __restrict__ buf,            // [B,V]
    const int* __restrict__ ids,        // [B,S]
    const float* __restrict__ attn,     // [B,S]
    const float* __restrict__ pgen)
{
    const int b = blockIdx.x;
    const float q = 1.0f - pgen[b];
#pragma unroll
    for (int r = 0; r < 2; ++r) {
        const int s = threadIdx.x + r * 256;
        const int id = ids[(size_t)b * S_ + s];
        atomicAdd(&buf[(size_t)b * V_ + id], q * attn[(size_t)b * S_ + s]);
    }
}

// ---------------------------------------------------------------------------
// k6: in-place log(x + EPS)
// ---------------------------------------------------------------------------
__global__ __launch_bounds__(256) void k6_log(float* __restrict__ buf)
{
    const size_t total = (size_t)B_ * V_;
    for (size_t i = (size_t)blockIdx.x * blockDim.x + threadIdx.x;
         i < total; i += (size_t)gridDim.x * blockDim.x) {
        buf[i] = __logf(buf[i] + EPS_);
    }
}

// ---------------------------------------------------------------------------
extern "C" void kernel_launch(void* const* d_in, const int* in_sizes, int n_in,
                              void* d_out, int out_size, void* d_ws, size_t ws_size,
                              hipStream_t stream) {
    const float* x        = (const float*)d_in[0];
    const float* embed    = (const float*)d_in[1];
    const float* hidden   = (const float*)d_in[2];
    const float* enc      = (const float*)d_in[3];
    const float* attn     = (const float*)d_in[4];
    const int*   ids      = (const int*)  d_in[5];
    const float* proj_w   = (const float*)d_in[6];
    const float* proj_b   = (const float*)d_in[7];
    const float* pgen_w   = (const float*)d_in[8];
    const float* pgen_b   = (const float*)d_in[9];
    float* out = (float*)d_out;

    float* ws    = (float*)d_ws;
    float* pgen  = ws;          // [B]
    float* row_m = ws + B_;     // [B]
    float* row_s = ws + 2 * B_; // [B]

    // 1) context + p_gen (fused, context never leaves LDS)
    k1_context_pgen<<<B_, 256, 0, stream>>>(enc, attn, hidden, embed,
                                            pgen_w, pgen_b, pgen);
    // 2) WMMA GEMM: logits -> d_out
    dim3 g2((V_ + 127) / 128, B_ / 16);
    k2_gemm_wmma<<<g2, 256, 0, stream>>>(x, proj_w, proj_b, out);
    // 3) softmax row stats
    k3_rowstats<<<B_, 256, 0, stream>>>(out, row_m, row_s);
    // 4) p_gen * softmax (in place)
    k4_mix<<<4096, 256, 0, stream>>>(out, pgen, row_m, row_s);
    // 5) pointer scatter (stream order = barrier between 4/5/6)
    k5_scatter<<<B_, 256, 0, stream>>>(out, ids, attn, pgen);
    // 6) log(x + eps) (in place)
    k6_log<<<4096, 256, 0, stream>>>(out);
}
